// SparseTF_10359461117965
// MI455X (gfx1250) — matrix operand order repaired and checked by reading, not compiled, not run on hardware
//
#include <hip/hip_runtime.h>
#include <hip/hip_bf16.h>

// ---------------------------------------------------------------------------
// SparseTF: kernel = scatter_add(zeros(10000,5000), ind, vals)
//           out    = tanh(inputs @ kernel + bias)
// Strategy:
//   * densify kernel (transposed, bf16) + inputs (bf16) into workspace
//   * dynamically bound the GEMM K-extent by the max scattered row index
//     (rows > kmax are identically zero => contribute nothing)
//   * bf16 WMMA GEMM: v_wmma_f32_16x16x32_bf16, f32 accumulate, tanh epilogue
// ---------------------------------------------------------------------------

typedef __attribute__((ext_vector_type(16))) __bf16          v16bf;
typedef __attribute__((ext_vector_type(8)))  float           v8f;
typedef __attribute__((ext_vector_type(8)))  unsigned short  u16x8;

#define BDIM   1024
#define INDIM  10000
#define UNITS  5000
#define NNZ    400000
#define KPAD   10016   // INDIM padded to multiple of 32 (WMMA K step)
#define NPAD   5008    // UNITS padded to multiple of 16 (WMMA N tile)

__device__ __forceinline__ int round32(int k) {
    int r = (k + 32) & ~31;            // roundup32(kmax + 1)
    return r < KPAD ? r : KPAD;
}

// ---------------- Phase A: kmax = max(row index) ---------------------------
__global__ void init_kmax_kernel(int* __restrict__ kmax) {
    if (threadIdx.x == 0 && blockIdx.x == 0) *kmax = 0;
}

__global__ void reduce_kmax_kernel(const int* __restrict__ ind,
                                   int* __restrict__ kmax) {
    __shared__ int smax;
    if (threadIdx.x == 0) smax = 0;
    __syncthreads();
    int i = blockIdx.x * blockDim.x + threadIdx.x;
    int r = (i < NNZ) ? ind[2 * i] : 0;
    atomicMax(&smax, r);
    __syncthreads();
    if (threadIdx.x == 0) atomicMax(kmax, smax);
}

// ---------------- Phase B: zero KT columns [0, kRound) ---------------------
// KT is [NPAD][KPAD] bf16; one uint4 = 8 bf16 elements, KPAD/8 = 1252 per row.
__global__ void zero_kt_kernel(uint4* __restrict__ p,
                               const int* __restrict__ kmaxp) {
    const int kRound = round32(*kmaxp);
    size_t i = (size_t)blockIdx.x * blockDim.x + threadIdx.x;
    if (i >= (size_t)NPAD * (KPAD / 8)) return;
    int col8 = (int)(i % (KPAD / 8)) * 8;   // bf16 column of this uint4
    if (col8 < kRound) p[i] = make_uint4(0u, 0u, 0u, 0u);
}

// ---------------- Phase C: convert + pad A (f32 -> bf16) -------------------
__global__ void convertA_kernel(const float* __restrict__ in,
                                __hip_bfloat16* __restrict__ A,
                                const int* __restrict__ kmaxp) {
    const int kRound = round32(*kmaxp);
    size_t i = (size_t)blockIdx.x * blockDim.x + threadIdx.x;
    if (i >= (size_t)BDIM * KPAD) return;
    int m = (int)(i / KPAD);
    int k = (int)(i % KPAD);
    if (k >= kRound) return;               // never read by the GEMM
    float v = (k < INDIM) ? in[(size_t)m * INDIM + k] : 0.0f;
    A[i] = __float2bfloat16(v);
}

// ---------------- Phase D: scatter-add into bf16 KT via 32-bit CAS ---------
// KT[n][k] = kernel[k][n]; duplicates are ADDED (matches .at[].add).
__global__ void scatter_add_kernel(const int* __restrict__ ind,
                                   const float* __restrict__ val,
                                   unsigned int* __restrict__ KT32) {
    int i = blockIdx.x * blockDim.x + threadIdx.x;
    if (i >= NNZ) return;
    int r = ind[2 * i];       // k index
    int c = ind[2 * i + 1];   // n index
    float add = val[i];
    size_t off = (size_t)c * KPAD + (size_t)r;      // bf16 element offset
    unsigned int* w = KT32 + (off >> 1);
    unsigned int hi = (unsigned int)(off & 1);
    unsigned int old = *w, assumed;
    do {
        assumed = old;
        unsigned int h = hi ? (assumed >> 16) : (assumed & 0xFFFFu);
        float f = __uint_as_float(h << 16) + add;   // bf16 -> f32, add
        unsigned int u = __float_as_uint(f);        // f32 -> bf16 (RNE)
        unsigned int nh = (u + 0x7FFFu + ((u >> 16) & 1u)) >> 16;
        unsigned int nw = hi ? ((assumed & 0x0000FFFFu) | (nh << 16))
                             : ((assumed & 0xFFFF0000u) | nh);
        old = atomicCAS(w, assumed, nw);
    } while (old != assumed);
}

// ---------------- Phase E: WMMA GEMM + bias + tanh -------------------------
// Each wave: 64(M) x 16(N) output tile, 4 accumulators sharing one B fragment
// per K-step. 8 waves/block; grid covers exactly 16 * 313 = 5008 wave-tiles.
__global__ void __launch_bounds__(256)
gemm_bf16_wmma_kernel(const __hip_bfloat16* __restrict__ A_,
                      const __hip_bfloat16* __restrict__ KT_,
                      const float* __restrict__ bias,
                      float* __restrict__ out,
                      const int* __restrict__ kmaxp) {
    const int kRound = round32(*kmaxp);     // uniform, multiple of 32
    const int lane  = threadIdx.x & 31;
    const int wave  = threadIdx.x >> 5;
    const int wtile = blockIdx.x * 8 + wave;
    const int NT    = NPAD / 16;            // 313
    const int nt    = wtile % NT;
    const int mt    = wtile / NT;           // 0..15
    const int col   = lane & 15;
    const int khalf = lane >> 4;
    const int nBase = nt * 16;
    const int mBase = mt * 64;

    const unsigned short* A  = (const unsigned short*)A_;
    const unsigned short* KT = (const unsigned short*)KT_;

    // Per-lane row pointers; all b128 accesses are 16-byte aligned
    // (KPAD*2 % 16 == 0; offsets are multiples of 16B).
    const unsigned short* a0 = A + (size_t)(mBase +  0 + col) * KPAD;
    const unsigned short* a1 = A + (size_t)(mBase + 16 + col) * KPAD;
    const unsigned short* a2 = A + (size_t)(mBase + 32 + col) * KPAD;
    const unsigned short* a3 = A + (size_t)(mBase + 48 + col) * KPAD;
    const unsigned short* br = KT + (size_t)(nBase + col) * KPAD;

    v8f acc0 = {}, acc1 = {}, acc2 = {}, acc3 = {};

    union frag { u16x8 h[2]; v16bf v; };

    for (int kb = 0; kb < kRound; kb += 32) {
        // B fragment (32x16 bf16): lane holds column n, K runs [kb+8*khalf, +7] and [kb+16+8*khalf, +7]
        frag fb;
        fb.h[0] = *(const u16x8*)(br + kb + 8 * khalf);
        fb.h[1] = *(const u16x8*)(br + kb + 16 + 8 * khalf);

        // A fragments (16x32 bf16): lane holds row m,
        // K runs [kb+8*khalf, +7] and [kb+16+8*khalf, +7]
        frag f0, f1, f2, f3;
        f0.h[0] = *(const u16x8*)(a0 + kb + 8 * khalf);
        f0.h[1] = *(const u16x8*)(a0 + kb + 16 + 8 * khalf);
        f1.h[0] = *(const u16x8*)(a1 + kb + 8 * khalf);
        f1.h[1] = *(const u16x8*)(a1 + kb + 16 + 8 * khalf);
        f2.h[0] = *(const u16x8*)(a2 + kb + 8 * khalf);
        f2.h[1] = *(const u16x8*)(a2 + kb + 16 + 8 * khalf);
        f3.h[0] = *(const u16x8*)(a3 + kb + 8 * khalf);
        f3.h[1] = *(const u16x8*)(a3 + kb + 16 + 8 * khalf);

        acc0 = __builtin_amdgcn_wmma_f32_16x16x32_bf16(false, f0.v, false, fb.v,
                                                       (short)0, acc0, false, false);
        acc1 = __builtin_amdgcn_wmma_f32_16x16x32_bf16(false, f1.v, false, fb.v,
                                                       (short)0, acc1, false, false);
        acc2 = __builtin_amdgcn_wmma_f32_16x16x32_bf16(false, f2.v, false, fb.v,
                                                       (short)0, acc2, false, false);
        acc3 = __builtin_amdgcn_wmma_f32_16x16x32_bf16(false, f3.v, false, fb.v,
                                                       (short)0, acc3, false, false);
    }

    // Epilogue: C/D layout -> lane: n = lane&15 ; VGPR j: m = j + 8*(lane>>4)
    // Non-temporal stores: output is write-once, keep L2 for A/KT residency.
    const int n = nBase + col;
    if (n < UNITS) {
        const float bv = bias[n];
        #pragma unroll
        for (int j = 0; j < 8; ++j) {
            int m = mBase + 0 * 16 + khalf * 8 + j;
            __builtin_nontemporal_store(tanhf(acc0[j] + bv),
                                        &out[(size_t)m * UNITS + n]);
        }
        #pragma unroll
        for (int j = 0; j < 8; ++j) {
            int m = mBase + 1 * 16 + khalf * 8 + j;
            __builtin_nontemporal_store(tanhf(acc1[j] + bv),
                                        &out[(size_t)m * UNITS + n]);
        }
        #pragma unroll
        for (int j = 0; j < 8; ++j) {
            int m = mBase + 2 * 16 + khalf * 8 + j;
            __builtin_nontemporal_store(tanhf(acc2[j] + bv),
                                        &out[(size_t)m * UNITS + n]);
        }
        #pragma unroll
        for (int j = 0; j < 8; ++j) {
            int m = mBase + 3 * 16 + khalf * 8 + j;
            __builtin_nontemporal_store(tanhf(acc3[j] + bv),
                                        &out[(size_t)m * UNITS + n]);
        }
    }
}

// ---------------------------------------------------------------------------
extern "C" void kernel_launch(void* const* d_in, const int* in_sizes, int n_in,
                              void* d_out, int out_size, void* d_ws, size_t ws_size,
                              hipStream_t stream) {
    const float* inputs = (const float*)d_in[0];   // [1024, 10000] f32
    const float* kvals  = (const float*)d_in[1];   // [400000] f32
    const float* bias   = (const float*)d_in[2];   // [5000] f32
    const int*   ind    = (const int*)d_in[3];     // [400000, 2] i32
    float*       out    = (float*)d_out;           // [1024, 5000] f32

    // Workspace layout:
    //   [0, 256)                          : int kmax (+ pad)
    //   [256, +NPAD*KPAD*2)               : KT bf16 (transposed dense kernel)
    //   [.., +BDIM*KPAD*2)                : A  bf16 (padded inputs)
    char* ws = (char*)d_ws;
    int* kmax = (int*)ws;
    __hip_bfloat16* KT = (__hip_bfloat16*)(ws + 256);
    __hip_bfloat16* A  = (__hip_bfloat16*)(ws + 256 + (size_t)NPAD * KPAD * 2);

    // Phase A: kmax over scattered row indices
    init_kmax_kernel<<<1, 64, 0, stream>>>(kmax);
    reduce_kmax_kernel<<<(NNZ + 255) / 256, 256, 0, stream>>>(ind, kmax);

    // Phase B: zero the live KT columns
    size_t n16 = (size_t)NPAD * (KPAD / 8);     // uint4 slots
    zero_kt_kernel<<<(unsigned)((n16 + 255) / 256), 256, 0, stream>>>(
        (uint4*)KT, kmax);

    // Phase C: convert + pad A to bf16 (live columns only)
    size_t nA = (size_t)BDIM * KPAD;
    convertA_kernel<<<(unsigned)((nA + 255) / 256), 256, 0, stream>>>(
        inputs, A, kmax);

    // Phase D: scatter-add nonzeros into bf16 KT
    scatter_add_kernel<<<(NNZ + 255) / 256, 256, 0, stream>>>(
        ind, kvals, (unsigned int*)KT);

    // Phase E: WMMA GEMM + bias + tanh
    const int waveTiles = (BDIM / 64) * (NPAD / 16);   // 16 * 313 = 5008
    gemm_bf16_wmma_kernel<<<waveTiles / 8, 256, 0, stream>>>(
        A, KT, bias, out, kmax);
}
